// sDNC_74345883893826
// MI455X (gfx1250) — matrix operand order repaired and checked by reading, not compiled, not run on hardware
//
#include <hip/hip_runtime.h>

// ---------------- types for WMMA (gfx1250, wave32) ----------------
typedef __bf16 bf16;
typedef __attribute__((ext_vector_type(16))) bf16  v16bf;
typedef __attribute__((ext_vector_type(8)))  bf16  v8bf;
typedef __attribute__((ext_vector_type(8)))  float v8f;

#define NWAVES 16          // 512 threads / wave32

// ---------------- problem dims ----------------
// B=8 L=512 H=128 C=256 N=32 W=512 R=8 HT=512 REL=97
// GEMV geometry (K padded to 32, N padded to 16):
#define K0CH 141   // layer0: [x(140)|reads(4096)|h0(256)] = 4492 -> 4512
#define NT0  64    // 1024 outputs
#define K1CH 16    // layers1/2: [h_prev_layer(256)|h_self(256)] = 512
#define KXICH 8    // xi: 256
#define NTXI 644   // 10299 -> 10304
#define KOCH 136   // out: [h(256)|reads(4096)] = 4352 exactly
#define NTO  9     // 140 -> 144
// bilinear GEMM geometry: K=148 -> 160 (5 chunks), N=148 -> 160 (10 tiles)
#define BKCH 5
#define BNT  10
#define BK_TILEHALF (BNT*BKCH*512)   // halves per k (25600)

// ---------------- LDS layout for the DNC kernel (bytes) ----------------
static constexpr unsigned LO_M     = 0;                    // 32*512 f32
static constexpr unsigned LO_READS = LO_M     + 32*512*4;  // 8*512 f32
static constexpr unsigned LO_XI    = LO_READS + 4096*4;    // 10304 f32
static constexpr unsigned LO_Z     = LO_XI    + 10304*4;   // 1024 f32
static constexpr unsigned LO_XBF   = LO_Z     + 1024*4;    // 4512 bf16 (16B aligned)
static constexpr unsigned LO_H0    = LO_XBF   + 4512*2;
static constexpr unsigned LO_C0    = LO_H0    + 256*4;
static constexpr unsigned LO_H1    = LO_C0    + 256*4;
static constexpr unsigned LO_C1    = LO_H1    + 256*4;
static constexpr unsigned LO_H2    = LO_C1    + 256*4;
static constexpr unsigned LO_C2    = LO_H2    + 256*4;
static constexpr unsigned LO_HCL   = LO_C2    + 256*4;
static constexpr unsigned LO_ER    = LO_HCL   + 256*4;     // 512 f32
static constexpr unsigned LO_WV    = LO_ER    + 512*4;
static constexpr unsigned LO_WM    = LO_WV    + 512*4;
static constexpr unsigned LO_RM    = LO_WM    + 512*4;     // 8*512 f32
static constexpr unsigned LO_WR    = LO_RM    + 4096*4;    // 8*32 f32
static constexpr unsigned LO_FW    = LO_WR    + 256*4;
static constexpr unsigned LO_BW    = LO_FW    + 256*4;
static constexpr unsigned LO_CRS   = LO_BW    + 256*4;
static constexpr unsigned LO_LNK   = LO_CRS   + 256*4;     // 32*32 f32
static constexpr unsigned LO_CW    = LO_LNK   + 1024*4;    // 32
static constexpr unsigned LO_AS    = LO_CW    + 32*4;
static constexpr unsigned LO_PSI   = LO_AS    + 32*4;
static constexpr unsigned LO_U     = LO_PSI   + 32*4;
static constexpr unsigned LO_WW    = LO_U     + 32*4;
static constexpr unsigned LO_WWN   = LO_WW    + 32*4;
static constexpr unsigned LO_P     = LO_WWN   + 32*4;
static constexpr unsigned LO_RB    = LO_P     + 32*4;      // 8
static constexpr unsigned LO_FG    = LO_RB    + 8*4;
static constexpr unsigned LO_SF    = LO_FG    + 8*4;
static constexpr unsigned LO_SB    = LO_SF    + 8*4;
static constexpr unsigned LO_PIS   = LO_SB    + 8*4;       // 24
static constexpr unsigned LO_FSUM  = LO_PIS   + 24*4;      // 8
static constexpr unsigned LO_BSUM  = LO_FSUM  + 8*4;
static constexpr unsigned LO_ORD   = LO_BSUM  + 8*4;       // 32 int
static constexpr unsigned LO_SCAL  = LO_ORD   + 32*4;      // wb, ga, gw, sum_ww
static constexpr unsigned LO_YBUF  = LO_SCAL  + 4*4;       // 144 f32
static constexpr unsigned LDS_TOTAL= LO_YBUF  + 144*4;     // ~172 KB (<320 KB WGP)

// ---------------- small device helpers ----------------
__device__ __forceinline__ unsigned short f2bf_bits(float f) {
  unsigned x = __float_as_uint(f);
  unsigned r = x + 0x7FFFu + ((x >> 16) & 1u);   // round-to-nearest-even
  return (unsigned short)(r >> 16);
}
__device__ __forceinline__ float sigm(float x) { return 1.f / (1.f + expf(-x)); }
__device__ __forceinline__ float softp(float x) { return (x > 20.f) ? x : log1pf(expf(x)); }

// y[nt*16+lane] = bias + sum_k x[k]*W[k][col], via v_wmma_f32_16x16x32_bf16.
// x replicated into all 16 A-rows -> every C row holds the same GEMV result.
__device__ __forceinline__ void wmma_gemv(
    const unsigned short* __restrict__ xbf,   // LDS bf16 vector, length kchunks*32
    const unsigned short* __restrict__ Wp,    // packed global bf16 B-tiles
    int kchunks, int ntiles,
    float* __restrict__ yout, const float* __restrict__ bias, int biasLen,
    int wave, int lane)
{
  const int koff = (lane >= 16) ? 8 : 0;
  for (int nt = wave; nt < ntiles; nt += NWAVES) {
    v8f c = {};
    const v16bf* bp = (const v16bf*)(const void*)(Wp + (size_t)nt * kchunks * 512);
    for (int kt = 0; kt < kchunks; ++kt) {
      const int kb = kt * 32 + koff;
      v8bf lo  = *(const v8bf*)(const void*)(xbf + kb);        // K {0..7}(+8)
      v8bf hi8 = *(const v8bf*)(const void*)(xbf + kb + 16);   // K {16..23}(+8)
      v16bf a = __builtin_shufflevector(lo, hi8, 0,1,2,3,4,5,6,7,8,9,10,11,12,13,14,15);
      // prefetch the weight stream a couple of tiles ahead (L2-resident, linear)
      __builtin_prefetch((const void*)(bp + kt * 32 + lane + 64), 0, 0);
      v16bf bm = bp[kt * 32 + lane];
      c = __builtin_amdgcn_wmma_f32_16x16x32_bf16(false, a, false, bm,
                                                  (short)0, c, false, false);
    }
    if (lane < 16) {
      const int col = nt * 16 + lane;
      float v = c[0];
      if (col < biasLen) v += bias[col];
      yout[col] = v;
    }
  }
}

__device__ __forceinline__ void lstm_cell(const float* z, int j, float* c, float* h) {
  float ig = sigm(z[j]);
  float fg = sigm(z[256 + j]);
  float g  = tanhf(z[512 + j]);
  float og = sigm(z[768 + j]);
  float cv = fg * c[j] + ig * g;
  c[j] = cv;
  h[j] = og * tanhf(cv);
}

// ---------------- weight packer: f32 [K x Nout] (optionally two stacked
// matrices) -> bf16 WMMA B-tiles [(nt*kchunks+kt)*32 + lane]*16 + half ----------
__global__ void pack_kernel(unsigned short* __restrict__ dst,
                            const float* __restrict__ W1, int K1,
                            const float* __restrict__ W2, int K2,
                            int Nout, int kchunks, int ntiles)
{
  long long total = (long long)ntiles * kchunks * 512;
  long long i = (long long)blockIdx.x * blockDim.x + threadIdx.x;
  if (i >= total) return;
  int h = (int)(i & 15);
  int l = (int)((i >> 4) & 31);
  long long rem = i >> 9;
  int kt = (int)(rem % kchunks);
  int nt = (int)(rem / kchunks);
  int col = nt * 16 + (l & 15);                 // lane -> column
  int k   = kt * 32 + (h & 7) + ((h >= 8) ? 16 : 0) + ((l >= 16) ? 8 : 0); // halves -> K, table order
  float v = 0.f;
  if (col < Nout) {
    if (k < K1)            v = W1[(size_t)k * Nout + col];
    else if (k < K1 + K2)  v = W2[(size_t)(k - K1) * Nout + col];
  }
  dst[i] = f2bf_bits(v);
}

// pack W_bili[k,i,j] transposed (rows=j as K, cols=i as N) into per-k B-tiles
__global__ void pack_bili_kernel(unsigned short* __restrict__ dst,
                                 const float* __restrict__ Wb)
{
  long long total = (long long)97 * BK_TILEHALF;
  long long i = (long long)blockIdx.x * blockDim.x + threadIdx.x;
  if (i >= total) return;
  int h = (int)(i & 15);
  int l = (int)((i >> 4) & 31);
  long long rem = i >> 9;
  int kt = (int)(rem % BKCH);
  int nt = (int)((rem / BKCH) % BNT);
  int k  = (int)(rem / (BKCH * BNT));
  int ci = nt * 16 + (l & 15);                   // N  = i index
  int rj = kt * 32 + (h & 7) + ((h >= 8) ? 16 : 0) + ((l >= 16) ? 8 : 0);   // K  = j index
  float v = 0.f;
  if (ci < 148 && rj < 148) v = Wb[(size_t)k * 148 * 148 + (size_t)ci * 148 + rj];
  dst[i] = f2bf_bits(v);
}

// f32 [4096 x 148] -> bf16 row-major padded [4096 x 160]
__global__ void bf16pad_kernel(const float* __restrict__ src, unsigned short* __restrict__ dst)
{
  int i = blockIdx.x * blockDim.x + threadIdx.x;
  if (i >= 4096 * 160) return;
  int row = i / 160, c = i % 160;
  dst[i] = (c < 148) ? f2bf_bits(src[(size_t)row * 148 + c]) : (unsigned short)0;
}

// ---------------- sent = concat(word_emb[idx], entity_emb[pos], ner_emb[ner]) --
__global__ void embed_kernel(const int* __restrict__ idx, const int* __restrict__ pos,
                             const int* __restrict__ ner,
                             const float* __restrict__ wemb, const float* __restrict__ eemb,
                             const float* __restrict__ nemb, float* __restrict__ sent)
{
  int i = blockIdx.x * blockDim.x + threadIdx.x;
  if (i >= 8 * 512 * 140) return;
  int f = i % 140, bl = i / 140;
  float v;
  if (f < 100)      v = wemb[(size_t)idx[bl] * 100 + f];
  else if (f < 120) v = eemb[(size_t)pos[bl] * 20 + (f - 100)];
  else              v = nemb[(size_t)ner[bl] * 20 + (f - 120)];
  sent[i] = v;
}

// ---------------- the DNC scan: one workgroup per batch element ----------------
__global__ void __launch_bounds__(512) dnc_kernel(
    const float* __restrict__ sent,
    const unsigned short* __restrict__ P0, const float* __restrict__ b0,
    const unsigned short* __restrict__ P1, const float* __restrict__ b1,
    const unsigned short* __restrict__ P2, const float* __restrict__ b2,
    const unsigned short* __restrict__ Pxi, const float* __restrict__ bxi,
    const unsigned short* __restrict__ Po,  const float* __restrict__ bo,
    float* __restrict__ dnc_y)
{
  extern __shared__ char smem[];
  float* M     = (float*)(smem + LO_M);
  float* reads = (float*)(smem + LO_READS);
  float* xiS   = (float*)(smem + LO_XI);
  float* z     = (float*)(smem + LO_Z);
  unsigned short* xbf = (unsigned short*)(smem + LO_XBF);
  float* h0 = (float*)(smem + LO_H0); float* c0 = (float*)(smem + LO_C0);
  float* h1 = (float*)(smem + LO_H1); float* c1 = (float*)(smem + LO_C1);
  float* h2 = (float*)(smem + LO_H2); float* c2 = (float*)(smem + LO_C2);
  float* hcl = (float*)(smem + LO_HCL);
  float* erS = (float*)(smem + LO_ER); float* wvS = (float*)(smem + LO_WV);
  float* wmS = (float*)(smem + LO_WM); float* rmS = (float*)(smem + LO_RM);
  float* wrS = (float*)(smem + LO_WR);
  float* fwS = (float*)(smem + LO_FW); float* bwS = (float*)(smem + LO_BW);
  float* crS = (float*)(smem + LO_CRS);
  float* Lnk = (float*)(smem + LO_LNK);
  float* cwS = (float*)(smem + LO_CW); float* aS = (float*)(smem + LO_AS);
  float* psiS= (float*)(smem + LO_PSI);float* uS = (float*)(smem + LO_U);
  float* wwS = (float*)(smem + LO_WW); float* wwN= (float*)(smem + LO_WWN);
  float* pS  = (float*)(smem + LO_P);
  float* rbS = (float*)(smem + LO_RB); float* fgS= (float*)(smem + LO_FG);
  float* sfS = (float*)(smem + LO_SF); float* sbS= (float*)(smem + LO_SB);
  float* piS = (float*)(smem + LO_PIS);
  float* fsum= (float*)(smem + LO_FSUM);float* bsum=(float*)(smem + LO_BSUM);
  int*   ordS= (int*)  (smem + LO_ORD);
  float* scal= (float*)(smem + LO_SCAL);
  float* ybuf= (float*)(smem + LO_YBUF);

  const int b = blockIdx.x;
  const int tid = threadIdx.x;
  const int lane = tid & 31, wave = tid >> 5;

  // ---- zero-init carry state ----
  for (int i = tid; i < 32 * 512; i += 512) M[i] = 0.f;
  for (int i = tid; i < 4096; i += 512) reads[i] = 0.f;
  if (tid < 256) { h0[tid]=c0[tid]=h1[tid]=c1[tid]=h2[tid]=c2[tid]=0.f; }
  if (tid < 32) {
    uS[tid]=wwS[tid]=pS[tid]=0.f;
    for (int r = 0; r < 8; ++r) wrS[r*32+tid] = 0.f;
  }
  for (int i = tid; i < 1024; i += 512) Lnk[i] = 0.f;
  __syncthreads();

  for (int t = 0; t < 512; ++t) {
    const float* xt = sent + ((size_t)b * 512 + t) * 140;

    // ---- layer0 input [x_t | reads | h0_prev], pad to 4512 ----
    for (int i = tid; i < K0CH * 32; i += 512) {
      float v = 0.f;
      if (i < 140)       v = xt[i];
      else if (i < 4236) v = reads[i - 140];
      else if (i < 4492) v = h0[i - 4236];
      xbf[i] = f2bf_bits(v);
    }
    __syncthreads();
    wmma_gemv(xbf, P0, K0CH, NT0, z, b0, 1024, wave, lane);
    __syncthreads();
    if (tid < 256) lstm_cell(z, tid, c0, h0);
    __syncthreads();

    // ---- layer1 ----
    { float v = (tid < 256) ? h0[tid] : h1[tid - 256]; xbf[tid] = f2bf_bits(v); }
    __syncthreads();
    wmma_gemv(xbf, P1, K1CH, NT0, z, b1, 1024, wave, lane);
    __syncthreads();
    if (tid < 256) lstm_cell(z, tid, c1, h1);
    __syncthreads();

    // ---- layer2 ----
    { float v = (tid < 256) ? h1[tid] : h2[tid - 256]; xbf[tid] = f2bf_bits(v); }
    __syncthreads();
    wmma_gemv(xbf, P2, K1CH, NT0, z, b2, 1024, wave, lane);
    __syncthreads();
    if (tid < 256) { lstm_cell(z, tid, c2, h2);
                     hcl[tid] = fminf(fmaxf(h2[tid], -20.f), 20.f); }
    __syncthreads();

    // ---- interface vector xi = clip(h2) @ W_xi + b_xi ----
    if (tid < 256) xbf[tid] = f2bf_bits(hcl[tid]);
    __syncthreads();
    wmma_gemv(xbf, Pxi, KXICH, NTXI, xiS, bxi, 10299, wave, lane);
    __syncthreads();

    // ---- interface transforms ----
    if (tid < 8) {
      rbS[tid] = 1.f + softp(xiS[4096 + tid]);
      fgS[tid] = sigm(xiS[5641 + tid]);
      sfS[tid] = 1.f + softp(xiS[10283 + tid]);
      sbS[tid] = 1.f + softp(xiS[10291 + tid]);
      float a0 = xiS[5651 + 3*tid], a1 = xiS[5652 + 3*tid], a2 = xiS[5653 + 3*tid];
      float mx = fmaxf(a0, fmaxf(a1, a2));
      float e0 = expf(a0-mx), e1 = expf(a1-mx), e2 = expf(a2-mx);
      float s = e0 + e1 + e2;
      piS[3*tid] = e0/s; piS[3*tid+1] = e1/s; piS[3*tid+2] = e2/s;
    }
    if (tid == 8)  scal[0] = 1.f + softp(xiS[4616]);  // wb
    if (tid == 9)  scal[1] = sigm(xiS[5649]);         // ga
    if (tid == 10) scal[2] = sigm(xiS[5650]);         // gw
    { int w = tid; // 512 threads == W
      erS[w] = sigm(xiS[4617 + w]);
      wvS[w] = xiS[5129 + w];
      wmS[w] = 0.1f + 0.9f * sigm(xiS[5675 + w]); }
    for (int i = tid; i < 4096; i += 512) rmS[i] = 0.1f + 0.9f * sigm(xiS[6187 + i]);
    __syncthreads();

    // ---- retention psi, usage u (uses OLD ww, OLD wr) ----
    if (tid < 32) {
      float ps = 1.f;
      for (int r = 0; r < 8; ++r) ps *= (1.f - fgS[r] * wrS[r*32 + tid]);
      psiS[tid] = ps;
      float uv = uS[tid], wv_ = wwS[tid];
      uS[tid] = (uv + wv_ - uv * wv_) * ps;
    }
    __syncthreads();

    // ---- allocation weighting (stable argsort of u ascending) ----
    if (tid < 32) {
      float uv = uS[tid]; int rk = 0;
      for (int j = 0; j < 32; ++j) {
        float uj = uS[j];
        rk += (uj < uv) || (uj == uv && j < tid);
      }
      ordS[rk] = tid;
    }
    __syncthreads();
    if (tid == 0) {
      float cp = 1.f;
      for (int k2 = 0; k2 < 32; ++k2) {
        int n = ordS[k2]; float us = uS[n];
        aS[n] = (1.f - us) * cp; cp *= us;
      }
    }
    // ---- write-head content addressing on OLD M (one pair per wave pass) ----
    for (int pidx = wave; pidx < 32; pidx += NWAVES) {
      const int n = pidx;
      float dot = 0.f, kn = 0.f, mn = 0.f;
      for (int w = lane; w < 512; w += 32) {
        float mk = wmS[w];
        float kw = xiS[4104 + w] * mk;
        float mw = M[n*512 + w] * mk;
        dot += kw * mw; kn += kw * kw; mn += mw * mw;
      }
      for (int o = 16; o > 0; o >>= 1) {
        dot += __shfl_down(dot, o, 32);
        kn  += __shfl_down(kn,  o, 32);
        mn  += __shfl_down(mn,  o, 32);
      }
      if (lane == 0) cwS[n] = dot / (sqrtf(kn) * sqrtf(mn) + 1e-6f);
    }
    __syncthreads();
    if (tid == 0) {   // softmax(beta*cos) over N
      float beta = scal[0], mx = -1e30f;
      for (int n = 0; n < 32; ++n) { float v = beta * cwS[n]; cwS[n] = v; mx = fmaxf(mx, v); }
      float s = 0.f;
      for (int n = 0; n < 32; ++n) { float e = expf(cwS[n] - mx); cwS[n] = e; s += e; }
      float inv = 1.f / s;
      for (int n = 0; n < 32; ++n) cwS[n] *= inv;
    }
    __syncthreads();
    if (tid < 32) {
      float ga = scal[1], gw = scal[2];
      wwN[tid] = gw * (ga * aS[tid] + (1.f - ga) * cwS[tid]);
    }
    __syncthreads();

    // ---- memory de-allocate + write ----
    for (int i = tid; i < 32 * 512; i += 512) {
      int n = i >> 9, w = i & 511;
      float m = M[i] * psiS[n];
      float wn = wwN[n];
      M[i] = m * (1.f - wn * erS[w]) + wn * wvS[w];
    }
    __syncthreads();

    // ---- read-head content addressing on NEW M ----
    for (int pidx = wave; pidx < 256; pidx += NWAVES) {
      const int r = pidx >> 5, n = pidx & 31;
      float dot = 0.f, kn = 0.f, mn = 0.f;
      for (int w = lane; w < 512; w += 32) {
        float mk = rmS[r*512 + w];
        float kw = xiS[r*512 + w] * mk;    // rk at xi offset 0
        float mw = M[n*512 + w] * mk;
        dot += kw * mw; kn += kw * kw; mn += mw * mw;
      }
      for (int o = 16; o > 0; o >>= 1) {
        dot += __shfl_down(dot, o, 32);
        kn  += __shfl_down(kn,  o, 32);
        mn  += __shfl_down(mn,  o, 32);
      }
      if (lane == 0) crS[r*32 + n] = dot / (sqrtf(kn) * sqrtf(mn) + 1e-6f);
    }
    __syncthreads();
    if (tid < 8) {    // per-head softmax over N
      float beta = rbS[tid], mx = -1e30f;
      for (int n = 0; n < 32; ++n) { float v = beta * crS[tid*32+n]; crS[tid*32+n] = v; mx = fmaxf(mx, v); }
      float s = 0.f;
      for (int n = 0; n < 32; ++n) { float e = expf(crS[tid*32+n] - mx); crS[tid*32+n] = e; s += e; }
      float inv = 1.f / s;
      for (int n = 0; n < 32; ++n) crS[tid*32+n] *= inv;
    }
    __syncthreads();

    // ---- temporal link matrix (uses OLD p) ----
    for (int i = tid; i < 1024; i += 512) {
      int ii = i >> 5, jj = i & 31;
      float L = (1.f - wwN[ii] - wwN[jj]) * Lnk[i] + wwN[ii] * pS[jj];
      Lnk[i] = (ii == jj) ? 0.f : L;
    }
    __syncthreads();
    if (tid == 0) { float s = 0.f; for (int n = 0; n < 32; ++n) s += wwN[n]; scal[3] = s; }
    __syncthreads();
    if (tid < 32) pS[tid] = (1.f - scal[3]) * pS[tid] + wwN[tid];
    // ---- forward/backward weights (use OLD wr), sharpen ----
    if (tid < 256) {
      int r = tid >> 5, i2 = tid & 31;
      float f = 0.f, bb = 0.f;
      for (int j = 0; j < 32; ++j) {
        float wj = wrS[r*32 + j];
        f  += Lnk[i2*32 + j] * wj;
        bb += Lnk[j*32 + i2] * wj;
      }
      fwS[tid] = powf(f + 1e-6f, sfS[r]);
      bwS[tid] = powf(bb + 1e-6f, sbS[r]);
    }
    __syncthreads();
    if (tid < 8) {
      float s1 = 0.f, s2 = 0.f;
      for (int i = 0; i < 32; ++i) { s1 += fwS[tid*32+i]; s2 += bwS[tid*32+i]; }
      fsum[tid] = s1; bsum[tid] = s2;
    }
    __syncthreads();
    // ---- new read weights ----
    if (tid < 256) {
      int r = tid >> 5;
      float fwv = fwS[tid] / fsum[r];
      float bwv = bwS[tid] / bsum[r];
      wrS[tid] = piS[3*r] * bwv + piS[3*r+1] * crS[tid] + piS[3*r+2] * fwv;
    }
    __syncthreads();
    // ---- reads = wr @ M ; carry ww ----
    for (int i = tid; i < 4096; i += 512) {
      int r = i >> 9, w = i & 511;
      float acc = 0.f;
      for (int n = 0; n < 32; ++n) acc += wrS[r*32 + n] * M[n*512 + w];
      reads[i] = acc;
    }
    if (tid < 32) wwS[tid] = wwN[tid];
    __syncthreads();

    // ---- y = [clip(h2) | reads] @ W_out + b_out ----
    for (int i = tid; i < KOCH * 32; i += 512) {
      float v = (i < 256) ? hcl[i] : reads[i - 256];
      xbf[i] = f2bf_bits(v);
    }
    __syncthreads();
    wmma_gemv(xbf, Po, KOCH, NTO, ybuf, bo, 140, wave, lane);
    __syncthreads();
    if (tid < 140) dnc_y[((size_t)b * 512 + t) * 140 + tid] = ybuf[tid];
    __syncthreads();
  }
}

// ---------------- post-scan: ctx = relu((y @ Wf + bf) @ Wr + br) ----------------
__global__ void final_kernel(const float* __restrict__ y,
                             const float* __restrict__ Wf, const float* __restrict__ bf,
                             const float* __restrict__ Wr, const float* __restrict__ br,
                             float* __restrict__ ctx)
{
  __shared__ float o140[140];
  __shared__ float f1[256];
  const int bl = blockIdx.x, tid = threadIdx.x;
  const float* yp = y + (size_t)bl * 140;
  if (tid < 140) o140[tid] = yp[tid];
  __syncthreads();
  float acc = bf[tid];
  for (int i = 0; i < 140; ++i) acc += o140[i] * Wf[i*256 + tid];
  f1[tid] = acc;
  __syncthreads();
  if (tid < 128) {
    float a = br[tid];
    for (int i = 0; i < 256; ++i) a += f1[i] * Wr[i*128 + tid];
    ctx[(size_t)bl * 128 + tid] = fmaxf(a, 0.f);
  }
}

// ---------------- s_rep / t_rep: mapping einsum + dis embedding ----------------
__global__ void maprep_kernel(const float* __restrict__ hm, const float* __restrict__ tm,
                              const float* __restrict__ ctx, const float* __restrict__ demb,
                              const int* __restrict__ dht, const int* __restrict__ dth,
                              float* __restrict__ srep, float* __restrict__ trep)
{
  const int bp = blockIdx.x;                  // b*512 + p
  const int bq = bp >> 9;
  const int h = threadIdx.x;                  // 0..127
  const float* hmp = hm + (size_t)bp * 512;
  const float* tmp = tm + (size_t)bp * 512;
  const float* cb  = ctx + (size_t)bq * 512 * 128;
  float s = 0.f, t2 = 0.f;
  for (int l = 0; l < 512; ++l) {
    float cv = cb[l*128 + h];
    s  += hmp[l] * cv;
    t2 += tmp[l] * cv;
  }
  float* sp = srep + (size_t)bp * 148;
  float* tp = trep + (size_t)bp * 148;
  sp[h] = s; tp[h] = t2;
  if (h < 20) {
    sp[128 + h] = demb[dht[bp] * 20 + h];
    tp[128 + h] = demb[dth[bp] * 20 + h];
  }
}

// ---------------- bilinear head via WMMA -------------------------------------
// out[bp,k] = sum_i s[bp,i] * (sum_j W[k,i,j] t[bp,j]) + b[k]
// One wave per (16-row tile mt, relation k): U = T_tile x Wk^T via
// v_wmma_f32_16x16x32_bf16 (5 K-chunks x 10 N-tiles), then fold s via the
// C-fragment layout (VGPR v <-> rows v / v+8, lane <-> column) + shfl tree.
__global__ void __launch_bounds__(32) bilinear_wmma_kernel(
    const unsigned short* __restrict__ Tbf,   // bf16 [4096 x 160]
    const float* __restrict__ srep,           // f32  [4096 x 148]
    const unsigned short* __restrict__ Pb,    // packed per-k B tiles
    const float* __restrict__ bb, float* __restrict__ out)
{
  const int mt = blockIdx.x / 97;             // 0..255 row tile
  const int k  = blockIdx.x % 97;
  const int lane = threadIdx.x;
  const int l15 = lane & 15;
  const int koff = (lane >= 16) ? 8 : 0;
  const int rowA = mt * 16 + l15;
  const v16bf* bp = (const v16bf*)(const void*)(Pb + (size_t)k * BK_TILEHALF);

  float acc[8];
  for (int v = 0; v < 8; ++v) acc[v] = 0.f;

  for (int nt = 0; nt < BNT; ++nt) {
    v8f c = {};
    for (int kt = 0; kt < BKCH; ++kt) {
      const unsigned short* ta = Tbf + (size_t)rowA * 160 + kt * 32 + koff;
      v8bf lo  = *(const v8bf*)(const void*)ta;
      v8bf hi8 = *(const v8bf*)(const void*)(ta + 16);
      v16bf a = __builtin_shufflevector(lo, hi8, 0,1,2,3,4,5,6,7,8,9,10,11,12,13,14,15);
      v16bf bm = bp[(nt * BKCH + kt) * 32 + lane];
      c = __builtin_amdgcn_wmma_f32_16x16x32_bf16(false, a, false, bm,
                                                  (short)0, c, false, false);
    }
    const int i = nt * 16 + l15;               // column of U
    if (i < 148) {
      const int rbase = mt * 16 + ((lane >= 16) ? 8 : 0);
      for (int v = 0; v < 8; ++v)
        acc[v] += c[v] * srep[(size_t)(rbase + v) * 148 + i];
    }
  }
  // reduce the 16 columns held per half-wave; lane0 -> rows v, lane16 -> rows v+8
  for (int v = 0; v < 8; ++v)
    for (int o = 8; o > 0; o >>= 1) acc[v] += __shfl_down(acc[v], o, 32);
  if (lane == 0 || lane == 16) {
    const int rbase = mt * 16 + ((lane == 16) ? 8 : 0);
    const float bk = bb[k];
    for (int v = 0; v < 8; ++v)
      out[(size_t)(rbase + v) * 97 + k] = acc[v] + bk;
  }
}

// ---------------- host launcher ----------------
extern "C" void kernel_launch(void* const* d_in, const int* in_sizes, int n_in,
                              void* d_out, int out_size, void* d_ws, size_t ws_size,
                              hipStream_t stream)
{
  (void)in_sizes; (void)n_in; (void)out_size; (void)ws_size;
  const int*   context_idxs = (const int*)  d_in[0];
  const int*   pos          = (const int*)  d_in[1];
  const int*   context_ner  = (const int*)  d_in[2];
  const float* h_mapping    = (const float*)d_in[5];
  const float* t_mapping    = (const float*)d_in[6];
  const int*   dis_h_2_t    = (const int*)  d_in[8];
  const int*   dis_t_2_h    = (const int*)  d_in[9];
  const float* word_emb     = (const float*)d_in[10];
  const float* entity_emb   = (const float*)d_in[11];
  const float* ner_emb      = (const float*)d_in[12];
  const float* dis_emb      = (const float*)d_in[13];
  const float* Wih0 = (const float*)d_in[14]; const float* Whh0 = (const float*)d_in[15];
  const float* b0   = (const float*)d_in[16];
  const float* Wih1 = (const float*)d_in[17]; const float* Whh1 = (const float*)d_in[18];
  const float* b1   = (const float*)d_in[19];
  const float* Wih2 = (const float*)d_in[20]; const float* Whh2 = (const float*)d_in[21];
  const float* b2   = (const float*)d_in[22];
  const float* W_xi = (const float*)d_in[23]; const float* b_xi = (const float*)d_in[24];
  const float* W_out= (const float*)d_in[25]; const float* b_out= (const float*)d_in[26];
  const float* W_fin= (const float*)d_in[27]; const float* b_fin= (const float*)d_in[28];
  const float* W_re = (const float*)d_in[29]; const float* b_re = (const float*)d_in[30];
  const float* W_bil= (const float*)d_in[31]; const float* b_bil= (const float*)d_in[32];

  char* ws = (char*)d_ws;
  size_t off = 0;
  auto carve = [&](size_t bytes) -> char* {
    char* p = ws + off;
    off += (bytes + 255) & ~(size_t)255;
    return p;
  };
  float* sent = (float*)carve((size_t)8*512*140*4);
  float* dncy = (float*)carve((size_t)8*512*140*4);
  float* ctx  = (float*)carve((size_t)8*512*128*4);
  float* srep = (float*)carve((size_t)8*512*148*4);
  float* trep = (float*)carve((size_t)8*512*148*4);
  unsigned short* P0  = (unsigned short*)carve((size_t)NT0 *K0CH *512*2);
  unsigned short* P1  = (unsigned short*)carve((size_t)NT0 *K1CH *512*2);
  unsigned short* P2  = (unsigned short*)carve((size_t)NT0 *K1CH *512*2);
  unsigned short* Pxi = (unsigned short*)carve((size_t)NTXI*KXICH*512*2);
  unsigned short* Po  = (unsigned short*)carve((size_t)NTO *KOCH *512*2);
  unsigned short* Pb  = (unsigned short*)carve((size_t)97*BK_TILEHALF*2);
  unsigned short* Tbf = (unsigned short*)carve((size_t)4096*160*2);

  embed_kernel<<<(8*512*140 + 255)/256, 256, 0, stream>>>(
      context_idxs, pos, context_ner, word_emb, entity_emb, ner_emb, sent);

  pack_kernel<<<((long long)NT0 *K0CH *512 + 255)/256, 256, 0, stream>>>(
      P0, Wih0, 4236, Whh0, 256, 1024, K0CH, NT0);
  pack_kernel<<<((long long)NT0 *K1CH *512 + 255)/256, 256, 0, stream>>>(
      P1, Wih1, 256, Whh1, 256, 1024, K1CH, NT0);
  pack_kernel<<<((long long)NT0 *K1CH *512 + 255)/256, 256, 0, stream>>>(
      P2, Wih2, 256, Whh2, 256, 1024, K1CH, NT0);
  pack_kernel<<<((long long)NTXI*KXICH*512 + 255)/256, 256, 0, stream>>>(
      Pxi, W_xi, 256, nullptr, 0, 10299, KXICH, NTXI);
  pack_kernel<<<((long long)NTO *KOCH *512 + 255)/256, 256, 0, stream>>>(
      Po, W_out, 4352, nullptr, 0, 140, KOCH, NTO);
  pack_bili_kernel<<<((long long)97*BK_TILEHALF + 255)/256, 256, 0, stream>>>(Pb, W_bil);

  dnc_kernel<<<8, 512, LDS_TOTAL, stream>>>(
      sent, P0, b0, P1, b1, P2, b2, Pxi, b_xi, Po, b_out, dncy);

  final_kernel<<<8*512, 256, 0, stream>>>(dncy, W_fin, b_fin, W_re, b_re, ctx);
  maprep_kernel<<<8*512, 128, 0, stream>>>(h_mapping, t_mapping, ctx, dis_emb,
                                           dis_h_2_t, dis_t_2_h, srep, trep);
  bf16pad_kernel<<<(4096*160 + 255)/256, 256, 0, stream>>>(trep, Tbf);
  bilinear_wmma_kernel<<<256*97, 32, 0, stream>>>(Tbf, srep, Pb, b_bil, (float*)d_out);
}
